// MassConservingNCA_515396075787
// MI455X (gfx1250) — compile-verified, matrix operand order checked by reading, not run on hardware
//
#include <hip/hip_runtime.h>
#include <hip/hip_bf16.h>

typedef __attribute__((ext_vector_type(16))) _Float16 v16h;
typedef __attribute__((ext_vector_type(8)))  float    v8f;

#define B_   4
#define C_   16
#define H_   512
#define W_   512
#define HID_ 128
#define NP   128      // pixels per row-tile per block
#define ROWS 4        // rows per block
#define PSTR 68       // percep LDS row stride in halves (64 + pad, even)
#define HSTR 132      // hidden LDS row stride in halves (128 + pad, even)

union FragH { v16h h; unsigned int u[8]; };

// --- tiny kernel: convert MLP weights to f16 in workspace ---
__global__ void nca_cvt_w(const float* __restrict__ w1, const float* __restrict__ w2,
                          _Float16* __restrict__ g1, _Float16* __restrict__ g2) {
    int i = blockIdx.x * 256 + threadIdx.x;
    if (i < HID_ * 64) g1[i] = (_Float16)w1[i];
    else if (i < HID_ * 64 + C_ * HID_) g2[i - HID_ * 64] = (_Float16)w2[i - HID_ * 64];
}

// --- main fused kernel: percep conv -> GEMM1(relu) -> GEMM2 -> Aff/T/normals ---
__global__ __launch_bounds__(256) void nca_mlp_kernel(
    const float* __restrict__ x, const float* __restrict__ rmask,
    const float* __restrict__ w1b,
    const _Float16* __restrict__ gW1, const _Float16* __restrict__ gW2,
    float* __restrict__ out, float* __restrict__ T)
{
    __shared__ _Float16 percepLDS[NP * PSTR];   // [pixel][k<64] f16
    __shared__ _Float16 hLDS[NP * HSTR];        // [pixel][k<128] f16
    __shared__ float rowbuf[2][3][136];         // 2 channels x 3 halo rows x 130 cols

    const int t     = threadIdx.x;
    const int lane  = t & 31;
    const int wv    = t >> 5;            // wave id 0..7
    const int khalf = (lane & 16) ? 8 : 0;
    const int lrow  = lane & 15;

    const int wt = blockIdx.x & 3;              // which 128-wide column strip
    const int hb = (blockIdx.x >> 2) & 127;     // which 4-row band
    const int b  = blockIdx.x >> 9;             // batch
    const int w0 = wt * NP;

    // Hoisted A-fragments (ISA 16-bit A layout: K = 2v + (v>=4?8:0) + (lane>=16?8:0))
    FragH a1[2];                     // GEMM1: this wave's 16 output rows, K=64 in 2 steps
#pragma unroll
    for (int ks = 0; ks < 2; ++ks)
#pragma unroll
        for (int v = 0; v < 8; ++v) {
            int k = ks * 32 + 2 * v + ((v & 4) << 1) + khalf;
            a1[ks].u[v] = *(const unsigned int*)(gW1 + (wv * 16 + lrow) * 64 + k);
        }
    FragH a2[4];                     // GEMM2: 16 output rows, K=128 in 4 steps
#pragma unroll
    for (int ks = 0; ks < 4; ++ks)
#pragma unroll
        for (int v = 0; v < 8; ++v) {
            int k = ks * 32 + 2 * v + ((v & 4) << 1) + khalf;
            a2[ks].u[v] = *(const unsigned int*)(gW2 + lrow * HID_ + k);
        }
    float bias[8];
#pragma unroll
    for (int r = 0; r < 8; ++r)
        bias[r] = w1b[wv * 16 + r + khalf];   // C/D row offset == khalf for lane>=16

    for (int hr = 0; hr < ROWS; ++hr) {
        const int row = hb * ROWS + hr;

        // ---------- stage 1: percep features (x, sobelX, sobelY, lap) -> LDS f16 ----------
        for (int c = 0; c < C_; c += 2) {
            __syncthreads();                           // rowbuf reuse guard
            for (int idx = t; idx < 780; idx += 256) { // 2 ch * 3 rows * 130 cols
                int cc = idx / 390, rem = idx % 390;
                int rr = rem / 130, j = rem % 130;
                int col = w0 - 1 + j; col += (col < 0) ? W_ : 0; col -= (col >= W_) ? W_ : 0;
                int rw = row - 1 + rr; rw += (rw < 0) ? H_ : 0; rw -= (rw >= H_) ? H_ : 0;
                rowbuf[cc][rr][j] = x[(((size_t)b * C_ + c + cc) * H_ + rw) * W_ + col];
            }
            __syncthreads();
            {
                int cc = t >> 7, p = t & 127;
                const float* r0 = rowbuf[cc][0] + p;
                const float* r1 = rowbuf[cc][1] + p;
                const float* r2 = rowbuf[cc][2] + p;
                float xc = r1[1];
                float gx = (r0[2] - r0[0]) + 2.f * (r1[2] - r1[0]) + (r2[2] - r2[0]);
                float gy = (r2[0] - r0[0]) + 2.f * (r2[1] - r0[1]) + (r2[2] - r0[2]);
                float lp = r0[0] + r0[2] + r2[0] + r2[2]
                         + 2.f * (r0[1] + r1[0] + r1[2] + r2[1]) - 12.f * xc;
                int ch = c + cc;
                _Float16* pr = percepLDS + p * PSTR;
                pr[ch]               = (_Float16)xc;
                pr[16 + ch * 3 + 0]  = (_Float16)gx;
                pr[16 + ch * 3 + 1]  = (_Float16)gy;
                pr[16 + ch * 3 + 2]  = (_Float16)lp;
            }
        }
        __syncthreads();

        // ---------- GEMM1: h(128 x 128pix) = W1 @ percep, +bias, relu ----------
        for (int nt = 0; nt < 8; ++nt) {
            int pix = nt * 16 + lrow;
            v8f acc = {};
#pragma unroll
            for (int ks = 0; ks < 2; ++ks) {
                FragH bf;
#pragma unroll
                for (int v = 0; v < 8; ++v) {
                    int k = ks * 32 + 2 * v + ((v & 4) << 1) + khalf;
                    bf.u[v] = *(const unsigned int*)(percepLDS + pix * PSTR + k);
                }
                acc = __builtin_amdgcn_wmma_f32_16x16x32_f16(
                          false, a1[ks].h, false, bf.h, (short)0, acc, false, false);
            }
#pragma unroll
            for (int r = 0; r < 8; ++r) {
                float hv = acc[r] + bias[r];
                hv = hv > 0.f ? hv : 0.f;
                hLDS[pix * HSTR + wv * 16 + r + khalf] = (_Float16)hv;
            }
        }
        __syncthreads();

        // ---------- GEMM2: y(16 x 128pix) = W2 @ h ----------
        int pix = wv * 16 + lrow;
        v8f y = {};
#pragma unroll
        for (int ks = 0; ks < 4; ++ks) {
            FragH bf;
#pragma unroll
            for (int v = 0; v < 8; ++v) {
                int k = ks * 32 + 2 * v + ((v & 4) << 1) + khalf;
                bf.u[v] = *(const unsigned int*)(hLDS + pix * HSTR + k);
            }
            y = __builtin_amdgcn_wmma_f32_16x16x32_f16(
                    false, a2[ks].h, false, bf.h, (short)0, y, false, false);
        }

        // ---------- epilogue: Aff/T + masked normal update ----------
        int wcol = w0 + pix;
        size_t pidx = ((size_t)b * H_ + row) * W_ + wcol;
        float rm = floorf(rmask[pidx] + 0.5f);
        if (khalf == 0) {            // lanes 0-15 hold channels 0..7
            float af0 = __expf(0.1f * y[0]);
            float af1 = __expf(0.1f * y[1]);
            float af2 = __expf(0.1f * y[2]);
            out[((size_t)(b * C_ + 0) * H_ + row) * W_ + wcol] = af0;  // Aff staged in out
            out[((size_t)(b * C_ + 1) * H_ + row) * W_ + wcol] = af1;
            out[((size_t)(b * C_ + 2) * H_ + row) * W_ + wcol] = af2;
            T[pidx] = af0 + af1 + af2;
#pragma unroll
            for (int r = 3; r < 8; ++r) {
                size_t gi = ((size_t)(b * C_ + r) * H_ + row) * W_ + wcol;
                out[gi] = x[gi] + y[r] * rm;
            }
        } else {                     // lanes 16-31 hold channels 8..15
#pragma unroll
            for (int r = 0; r < 8; ++r) {
                size_t gi = ((size_t)(b * C_ + r + 8) * H_ + row) * W_ + wcol;
                out[gi] = x[gi] + y[r] * rm;
            }
        }
    }
}

// --- E = box3x3_wrap(T);  U = (x0+x1+x2)/E ---
__global__ __launch_bounds__(256) void nca_E_kernel(
    const float* __restrict__ x, const float* __restrict__ T, float* __restrict__ U)
{
    int idx = blockIdx.x * 256 + threadIdx.x;
    if (idx >= B_ * H_ * W_) return;
    int w = idx % W_;
    int h = (idx / W_) % H_;
    int b = idx / (W_ * H_);
    float e = 0.f;
#pragma unroll
    for (int dh = -1; dh <= 1; ++dh) {
        int hh = h + dh; hh += (hh < 0) ? H_ : 0; hh -= (hh >= H_) ? H_ : 0;
#pragma unroll
        for (int dw = -1; dw <= 1; ++dw) {
            int ww = w + dw; ww += (ww < 0) ? W_ : 0; ww -= (ww >= W_) ? W_ : 0;
            e += T[((size_t)b * H_ + hh) * W_ + ww];
        }
    }
    size_t base = ((size_t)b * C_ * H_ + h) * W_ + w;
    float sx = x[base] + x[base + (size_t)H_ * W_] + x[base + 2 * (size_t)H_ * W_];
    U[idx] = sx / e;
}

// --- R = box3x3_wrap(U);  out[:, :3] = Aff * R  (Aff staged in out, in-place) ---
__global__ __launch_bounds__(256) void nca_R_kernel(
    const float* __restrict__ U, float* __restrict__ out)
{
    int idx = blockIdx.x * 256 + threadIdx.x;
    if (idx >= B_ * H_ * W_) return;
    int w = idx % W_;
    int h = (idx / W_) % H_;
    int b = idx / (W_ * H_);
    float r = 0.f;
#pragma unroll
    for (int dh = -1; dh <= 1; ++dh) {
        int hh = h + dh; hh += (hh < 0) ? H_ : 0; hh -= (hh >= H_) ? H_ : 0;
#pragma unroll
        for (int dw = -1; dw <= 1; ++dw) {
            int ww = w + dw; ww += (ww < 0) ? W_ : 0; ww -= (ww >= W_) ? W_ : 0;
            r += U[((size_t)b * H_ + hh) * W_ + ww];
        }
    }
#pragma unroll
    for (int c = 0; c < 3; ++c) {
        size_t gi = ((size_t)(b * C_ + c) * H_ + h) * W_ + w;
        out[gi] = out[gi] * r;
    }
}

extern "C" void kernel_launch(void* const* d_in, const int* in_sizes, int n_in,
                              void* d_out, int out_size, void* d_ws, size_t ws_size,
                              hipStream_t stream) {
    const float* x    = (const float*)d_in[0];
    const float* rm   = (const float*)d_in[1];
    const float* w1w  = (const float*)d_in[2];
    const float* w1b  = (const float*)d_in[3];
    const float* w2w  = (const float*)d_in[4];
    float* out = (float*)d_out;
    char*  ws  = (char*)d_ws;

    _Float16* gW1 = (_Float16*)ws;                       // 128*64  f16 = 16384 B
    _Float16* gW2 = (_Float16*)(ws + 16384);             // 16*128  f16 =  4096 B
    float*    T   = (float*)(ws + 20480);                // B*H*W f32 = 4 MiB
    float*    U   = (float*)(ws + 20480 + sizeof(float) * (size_t)B_ * H_ * W_);

    nca_cvt_w<<<40, 256, 0, stream>>>(w1w, w2w, gW1, gW2);

    const int grid_mlp = B_ * (H_ / ROWS) * (W_ / NP);   // 4 * 128 * 4 = 2048
    nca_mlp_kernel<<<grid_mlp, 256, 0, stream>>>(x, rm, w1b, gW1, gW2, out, T);

    const int grid_px = (B_ * H_ * W_ + 255) / 256;      // 4096
    nca_E_kernel<<<grid_px, 256, 0, stream>>>(x, T, U);
    nca_R_kernel<<<grid_px, 256, 0, stream>>>(U, out);
}
